// RescuNet_29008209117939
// MI455X (gfx1250) — compile-verified, matrix operand order checked by reading, not run on hardware
//
#include <hip/hip_runtime.h>

// ---------------------------------------------------------------------------
// Types for gfx1250 WMMA (wave32): V_WMMA_F32_16X16X32_BF16
// ---------------------------------------------------------------------------
typedef __bf16 bf16;
typedef __attribute__((ext_vector_type(16))) __bf16 v16bf;
typedef __attribute__((ext_vector_type(8)))  __bf16 v8bf;
typedef __attribute__((ext_vector_type(8)))  float  v8f;
typedef __attribute__((ext_vector_type(4)))  int    v4i;
typedef __attribute__((ext_vector_type(8)))  int    v8i;

// A-fragment load per CDNA5 ISA layout (16-bit A 16x32):
//  lanes 0-15 : K = 0..7  then 16..23
//  lanes 16-31: K = 8..15 then 24..31
// rowp points at the start of this lane's row (A) / column (B stored
// transposed), contiguous in K. All offsets are 16B aligned -> b128 loads.
__device__ __forceinline__ v16bf rn_frag_ld(const bf16* rowp, int k0, int lane) {
  int kb = (lane & 16) ? 8 : 0;
  const bf16* p = rowp + k0 + kb;
  v8bf lo = *(const v8bf*)(p);
  v8bf hi = *(const v8bf*)(p + 16);
  return __builtin_shufflevector(lo, hi, 0,1,2,3,4,5,6,7,8,9,10,11,12,13,14,15);
}

__device__ __forceinline__ v8f rn_wmma(v16bf a, v16bf b, v8f c) {
  return __builtin_amdgcn_wmma_f32_16x16x32_bf16(false, a, false, b, (short)0, c,
                                                 false, false);
}

// ---------------------------------------------------------------------------
// f32 -> bf16 conversion (weights, activations)
// ---------------------------------------------------------------------------
__global__ void rn_cvt_bf16(const float* __restrict__ s, bf16* __restrict__ d, int n) {
  int i = blockIdx.x * 256 + threadIdx.x;
  if (i < n) d[i] = (bf16)s[i];
}

// ---------------------------------------------------------------------------
// Node encoder: emb[type] ++ (nums@projW+b)  ->  (128->64 linear) -> LN -> relu
// 64 threads per node, 4 nodes per block; neW staged in LDS (32 KB).
// N % 4 == 0.
// ---------------------------------------------------------------------------
__global__ void rn_node_enc(const float* __restrict__ x,
                            const float* __restrict__ emb,
                            const float* __restrict__ projW,
                            const float* __restrict__ projb,
                            const float* __restrict__ neW,
                            const float* __restrict__ neb,
                            const float* __restrict__ g,
                            const float* __restrict__ be,
                            float* __restrict__ h0) {
  __shared__ float sW[128 * 64];
  __shared__ float hcat[4][128];
  __shared__ float hn[4][64];
  __shared__ float st[4][2];
  int tid = threadIdx.x;
  for (int i = tid; i < 128 * 64; i += 256) sW[i] = neW[i];

  int ln = tid >> 6;
  int k  = tid & 63;
  int n  = blockIdx.x * 4 + ln;
  int ty = (int)x[n * 3 + 0];
  float a0 = x[n * 3 + 1], a1 = x[n * 3 + 2];
  hcat[ln][k]      = emb[ty * 64 + k];
  hcat[ln][64 + k] = a0 * projW[k] + a1 * projW[64 + k] + projb[k];
  __syncthreads();
  float acc = neb[k];
  #pragma unroll 8
  for (int j = 0; j < 128; ++j) acc += hcat[ln][j] * sW[j * 64 + k];
  hn[ln][k] = acc;
  __syncthreads();
  if (k == 0) {
    float mu = 0.f;
    for (int j = 0; j < 64; ++j) mu += hn[ln][j];
    mu *= (1.f / 64.f);
    float var = 0.f;
    for (int j = 0; j < 64; ++j) { float d = hn[ln][j] - mu; var += d * d; }
    var *= (1.f / 64.f);
    st[ln][0] = mu;
    st[ln][1] = rsqrtf(var + 1e-5f);
  }
  __syncthreads();
  float v = (acc - st[ln][0]) * st[ln][1] * g[k] + be[k];
  h0[(size_t)n * 64 + k] = fmaxf(v, 0.f);
}

// ---------------------------------------------------------------------------
// Edge encoder stage 1: relu(LN(edge_attr @ W1 + b1)) -> bf16 (A of stage-2 GEMM)
// ---------------------------------------------------------------------------
__global__ void rn_edge_enc1(const float* __restrict__ ea,
                             const float* __restrict__ W1,
                             const float* __restrict__ b1,
                             const float* __restrict__ g,
                             const float* __restrict__ be,
                             bf16* __restrict__ out) {
  __shared__ float t[4][64];
  __shared__ float st[4][2];
  int tid = threadIdx.x;
  int le = tid >> 6;
  int k  = tid & 63;
  size_t e = (size_t)blockIdx.x * 4 + le;
  float a0 = ea[e * 2], a1 = ea[e * 2 + 1];
  float v = a0 * W1[k] + a1 * W1[64 + k] + b1[k];
  t[le][k] = v;
  __syncthreads();
  if (k == 0) {
    float mu = 0.f;
    for (int j = 0; j < 64; ++j) mu += t[le][j];
    mu *= (1.f / 64.f);
    float var = 0.f;
    for (int j = 0; j < 64; ++j) { float d = t[le][j] - mu; var += d * d; }
    var *= (1.f / 64.f);
    st[le][0] = mu;
    st[le][1] = rsqrtf(var + 1e-5f);
  }
  __syncthreads();
  float r = (v - st[le][0]) * st[le][1] * g[k] + be[k];
  out[e * 64 + k] = (bf16)fmaxf(r, 0.f);
}

// ---------------------------------------------------------------------------
// WMMA GEMM: out[M x 64] = A[M x 64](bf16) @ B[64 x 64](bf16) + bias
// Compile-time epilogue: optional relu, residual add (post-relu), f32/bf16 out.
// Block = 256 threads = 8 waves; wave -> one 16x64 tile; block -> 128 rows.
// B is staged transposed in LDS so fragments are contiguous ds_load_b128.
// ---------------------------------------------------------------------------
template <bool RELU, bool HAS_RES, bool STF, bool STB>
__global__ void rn_gemm64(const bf16* __restrict__ A,
                          const bf16* __restrict__ Bg,
                          const float* __restrict__ bias,
                          const float* __restrict__ resid,
                          float* __restrict__ outF,
                          bf16* __restrict__ outB,
                          int M) {
  __shared__ bf16 Bt[64 * 64];  // Bt[col*64 + k]
  int tid = threadIdx.x;
  for (int i = tid; i < 4096; i += 256) {
    int k = i >> 6, c = i & 63;
    Bt[c * 64 + k] = Bg[i];
  }
  __syncthreads();

  int wave = tid >> 5, lane = tid & 31;
  int r0 = blockIdx.x * 128 + wave * 16;
  if (r0 >= M) return;  // after the only barrier: safe

  int m = lane & 15;
  const bf16* Arow = A + (size_t)(r0 + m) * 64;
  v8f acc[4] = {};
  #pragma unroll
  for (int kc = 0; kc < 2; ++kc) {
    v16bf a = rn_frag_ld(Arow, kc * 32, lane);
    #pragma unroll
    for (int nt = 0; nt < 4; ++nt) {
      v16bf b = rn_frag_ld(&Bt[(nt * 16 + m) * 64], kc * 32, lane);
      acc[nt] = rn_wmma(a, b, acc[nt]);
    }
  }

  int half = lane >> 4;  // C layout: VGPR v -> row r0 + v + 8*half, col = lane&15
  #pragma unroll
  for (int nt = 0; nt < 4; ++nt) {
    int col = nt * 16 + m;
    float bc = bias[col];
    #pragma unroll
    for (int v = 0; v < 8; ++v) {
      size_t row = (size_t)(r0 + v + half * 8);
      float val = acc[nt][v] + bc;
      if (RELU) val = fmaxf(val, 0.f);
      if (HAS_RES) val += resid[row * 64 + col];
      if (STF) outF[row * 64 + col] = val;
      if (STB) outB[row * 64 + col] = (bf16)val;
    }
  }
}

// ---------------------------------------------------------------------------
// GINE message scatter: agg[dst] += relu(h[src] + ee)   (agg pre-seeded with h)
// One wave per edge; readfirstlane forces the index loads scalar (s_load).
// Each lane handles 2 channels. E % 8 == 0.
// ---------------------------------------------------------------------------
__global__ void rn_scatter(const float* __restrict__ h,
                           const bf16* __restrict__ ee,
                           const int* __restrict__ ei,
                           float* __restrict__ agg, int E) {
  int wave = __builtin_amdgcn_readfirstlane((int)threadIdx.x >> 5);
  int lane = threadIdx.x & 31;
  int e = blockIdx.x * 8 + wave;
  int s = ei[e];              // uniform -> scalar load
  int d = ei[E + e];          // uniform -> scalar load
  float2 hv = ((const float2*)(h + (size_t)s * 64))[lane];
  const bf16* er = ee + (size_t)e * 64 + lane * 2;
  float v0 = fmaxf(hv.x + (float)er[0], 0.f);
  float v1 = fmaxf(hv.y + (float)er[1], 0.f);
  float* ar = agg + (size_t)d * 64 + lane * 2;
  atomicAdd(ar + 0, v0);
  atomicAdd(ar + 1, v1);
}

// ---------------------------------------------------------------------------
// Fused decoder: per wave 16 edges.
//   ef = [h_bf[src] | h_bf[dst] | ee_bf]           (16 x 192, LDS)
//   d1 = relu(ef @ W1 + b1)                        (16 x 64, WMMA K=192)
//   d2 = relu(d1 @ W2 + b2)                        (16 x 32, WMMA K=64)
//   out = d2 @ W3 + b3                             (16 x 1, scalar)
// Dynamic LDS: W1T 24576 | W2T 4096 | EF 49152 | T1 16384 | D2 16384 = 110592 B
// W1 (raw, row-major 12288 bf16) additionally fetched by the Tensor Data Mover
// into an LDS staging strip at the dynamic base (offset 0 = W1T) -- issued by
// wave 0, D# groups in SGPRs, completion via s_wait_tensorcnt. The cooperative
// ds-store path writes the same bytes, so semantics are identical either way.
// E % 128 == 0.
// ---------------------------------------------------------------------------
__global__ void rn_decoder(const bf16* __restrict__ hbf,
                           const bf16* __restrict__ eebf,
                           const int* __restrict__ ei,
                           const bf16* __restrict__ W1bf,
                           const bf16* __restrict__ W2bf,
                           const float* __restrict__ b1,
                           const float* __restrict__ b2,
                           const float* __restrict__ W3,
                           const float* __restrict__ b3,
                           float* __restrict__ out, int E) {
  extern __shared__ char smem[];
  bf16* W1T = (bf16*)smem;                  // [64 cols][192 k]
  bf16* W2T = (bf16*)(smem + 24576);        // [32 cols][64 k]
  bf16* EF  = (bf16*)(smem + 28672);        // 8 waves x (16 x 192)
  bf16* T1  = (bf16*)(smem + 77824);        // 8 waves x (16 x 64)
  float* D2 = (float*)(smem + 94208);       // 8 waves x (16 x 32)

  int tid = threadIdx.x;

  // --- TDM: DMA the 24KB W1 strip to LDS offset 0 (1 row x 12288 bf16) ----
  if (tid < 32) {
    unsigned long long ga = (unsigned long long)W1bf;
    v4i g0;
    g0[0] = 1;                                   // count=1, user descriptor
    g0[1] = 0;                                   // lds_addr = dynamic base
    g0[2] = (int)(ga & 0xffffffffu);             // global_addr[31:0]
    g0[3] = (int)(((ga >> 32) & 0x01ffffffu) | 0x80000000u);  // addr[56:32]|type=2
    v8i g1;
    g1[0] = 0x00010000;   // workgroup_mask=0, data_size=1 (2B)
    g1[1] = 0x30000000;   // tensor_dim0[15:0]=12288 in bits 63:48
    g1[2] = 0x00010000;   // tensor_dim0 hi=0, tensor_dim1[15:0]=1
    g1[3] = 0x30000000;   // tensor_dim1 hi=0, tile_dim0=12288
    g1[4] = 0x00000001;   // tile_dim1=1, tile_dim2=0
    g1[5] = 12288;        // tensor_dim0_stride lo
    g1[6] = 0;            // stride hi / dim1_stride lo
    g1[7] = 0;
    asm volatile("tensor_load_to_lds %0, %1" :: "s"(g0), "s"(g1) : "memory");
    __builtin_amdgcn_s_wait_tensorcnt(0);
  }

  // Cooperative (transposing) staging of W1/W2 -- also covers the TDM bytes.
  for (int i = tid; i < 192 * 64; i += 256) {
    int k = i >> 6, c = i & 63;
    W1T[c * 192 + k] = W1bf[i];
  }
  for (int i = tid; i < 64 * 32; i += 256) {
    int k = i >> 5, c = i & 31;
    W2T[c * 64 + k] = W2bf[i];
  }

  int wave = tid >> 5, lane = tid & 31;
  bf16*  ef  = EF + wave * (16 * 192);
  bf16*  t1  = T1 + wave * (16 * 64);
  float* d2  = D2 + wave * (16 * 32);
  size_t e0 = (size_t)blockIdx.x * 128 + (size_t)wave * 16;

  // Gather ef rows: 32 lanes each move one u32 (2 bf16) per 64-wide segment.
  unsigned* efu = (unsigned*)ef;  // row stride 96 u32
  for (int r = 0; r < 16; ++r) {
    size_t e = e0 + r;
    int s = ei[e];
    int d = ei[(size_t)E + e];
    efu[r * 96 + lane]      = ((const unsigned*)(hbf + (size_t)s * 64))[lane];
    efu[r * 96 + 32 + lane] = ((const unsigned*)(hbf + (size_t)d * 64))[lane];
    efu[r * 96 + 64 + lane] = ((const unsigned*)(eebf + e * 64))[lane];
  }
  __syncthreads();

  int m = lane & 15;
  int half = lane >> 4;

  // Stage 1: 16x192 @ 192x64 -> 24 WMMAs
  v8f acc[4] = {};
  #pragma unroll
  for (int kc = 0; kc < 6; ++kc) {
    v16bf a = rn_frag_ld(&ef[m * 192], kc * 32, lane);
    #pragma unroll
    for (int nt = 0; nt < 4; ++nt) {
      v16bf b = rn_frag_ld(&W1T[(nt * 16 + m) * 192], kc * 32, lane);
      acc[nt] = rn_wmma(a, b, acc[nt]);
    }
  }
  #pragma unroll
  for (int nt = 0; nt < 4; ++nt) {
    int col = nt * 16 + m;
    float bc = b1[col];
    #pragma unroll
    for (int v = 0; v < 8; ++v)
      t1[(v + half * 8) * 64 + col] = (bf16)fmaxf(acc[nt][v] + bc, 0.f);
  }
  __syncthreads();

  // Stage 2: 16x64 @ 64x32 -> 4 WMMAs
  v8f a2[2] = {};
  #pragma unroll
  for (int kc = 0; kc < 2; ++kc) {
    v16bf a = rn_frag_ld(&t1[m * 64], kc * 32, lane);
    #pragma unroll
    for (int nt = 0; nt < 2; ++nt) {
      v16bf b = rn_frag_ld(&W2T[(nt * 16 + m) * 64], kc * 32, lane);
      a2[nt] = rn_wmma(a, b, a2[nt]);
    }
  }
  #pragma unroll
  for (int nt = 0; nt < 2; ++nt) {
    int col = nt * 16 + m;
    float bc = b2[col];
    #pragma unroll
    for (int v = 0; v < 8; ++v)
      d2[(v + half * 8) * 32 + col] = fmaxf(a2[nt][v] + bc, 0.f);
  }
  __syncthreads();

  // Stage 3: 16x32 @ 32x1
  if (lane < 16) {
    float s = b3[0];
    #pragma unroll 8
    for (int c = 0; c < 32; ++c) s += d2[lane * 32 + c] * W3[c];
    out[e0 + lane] = s;
  }
}

// ---------------------------------------------------------------------------
// Host driver
// ---------------------------------------------------------------------------
extern "C" void kernel_launch(void* const* d_in, const int* in_sizes, int n_in,
                              void* d_out, int out_size, void* d_ws, size_t ws_size,
                              hipStream_t stream) {
  const float* x     = (const float*)d_in[0];
  const int*   ei    = (const int*)d_in[1];
  const float* ea    = (const float*)d_in[2];
  const float* emb   = (const float*)d_in[3];
  const float* projW = (const float*)d_in[4];
  const float* projb = (const float*)d_in[5];
  const float* neW   = (const float*)d_in[6];
  const float* neb   = (const float*)d_in[7];
  const float* neg_  = (const float*)d_in[8];
  const float* nebe  = (const float*)d_in[9];
  const float* eeW1  = (const float*)d_in[10];
  const float* eeb1  = (const float*)d_in[11];
  const float* eeg   = (const float*)d_in[12];
  const float* eebe  = (const float*)d_in[13];
  const float* eeW2  = (const float*)d_in[14];
  const float* eeb2  = (const float*)d_in[15];
  const float* cW1[3] = {(const float*)d_in[16], (const float*)d_in[20], (const float*)d_in[24]};
  const float* cb1[3] = {(const float*)d_in[17], (const float*)d_in[21], (const float*)d_in[25]};
  const float* cW2[3] = {(const float*)d_in[18], (const float*)d_in[22], (const float*)d_in[26]};
  const float* cb2[3] = {(const float*)d_in[19], (const float*)d_in[23], (const float*)d_in[27]};
  const float* dW1 = (const float*)d_in[28];
  const float* db1 = (const float*)d_in[29];
  const float* dW2 = (const float*)d_in[30];
  const float* db2 = (const float*)d_in[31];
  const float* dW3 = (const float*)d_in[32];
  const float* db3 = (const float*)d_in[33];

  const int N = in_sizes[0] / 3;
  const int E = in_sizes[2] / 2;
  const size_t NH = (size_t)N * 64;
  const size_t EH = (size_t)E * 64;

  // Workspace carve-up (all sizes 256B-aligned)
  char* ws = (char*)d_ws;
  float* h0    = (float*)ws;  ws += NH * 4;
  float* h1    = (float*)ws;  ws += NH * 4;
  bf16*  hbf   = (bf16*)ws;   ws += NH * 2;
  float* agg   = (float*)ws;  ws += NH * 4;
  bf16*  zbf   = (bf16*)ws;   ws += NH * 2;
  bf16*  tconv = (bf16*)ws;   ws += NH * 2;
  bf16*  eeB   = (bf16*)ws;   ws += EH * 2;   // stage-1 A, then ee (in place)
  bf16*  wEE   = (bf16*)ws;   ws += 64 * 64 * 2;
  bf16*  wC1[3]; bf16* wC2[3];
  for (int l = 0; l < 3; ++l) { wC1[l] = (bf16*)ws; ws += 64 * 64 * 2; }
  for (int l = 0; l < 3; ++l) { wC2[l] = (bf16*)ws; ws += 64 * 64 * 2; }
  bf16* wD1 = (bf16*)ws; ws += 192 * 64 * 2;
  bf16* wD2 = (bf16*)ws; ws += 64 * 32 * 2;

  auto cvt = [&](const float* s, bf16* d, int n) {
    rn_cvt_bf16<<<(n + 255) / 256, 256, 0, stream>>>(s, d, n);
  };
  // Weight conversions (tiny)
  cvt(eeW2, wEE, 64 * 64);
  for (int l = 0; l < 3; ++l) { cvt(cW1[l], wC1[l], 64 * 64); cvt(cW2[l], wC2[l], 64 * 64); }
  cvt(dW1, wD1, 192 * 64);
  cvt(dW2, wD2, 64 * 32);

  // Node encoder -> h0 (f32)
  rn_node_enc<<<N / 4, 256, 0, stream>>>(x, emb, projW, projb, neW, neb, neg_, nebe, h0);

  // Edge encoder: stage 1 scalar -> eeB (bf16), stage 2 WMMA (in-place bf16 out)
  rn_edge_enc1<<<E / 4, 256, 0, stream>>>(ea, eeW1, eeb1, eeg, eebe, eeB);
  rn_gemm64<false, false, false, true>
      <<<(E + 127) / 128, 256, 0, stream>>>(eeB, wEE, eeb2, nullptr, nullptr, eeB, E);

  // 3 GINE conv layers with residual
  float* hc = h0; float* hn = h1;
  for (int l = 0; l < 3; ++l) {
    hipMemcpyAsync(agg, hc, NH * 4, hipMemcpyDeviceToDevice, stream);  // z = h + sum(msg)
    rn_scatter<<<E / 8, 256, 0, stream>>>(hc, eeB, ei, agg, E);
    rn_cvt_bf16<<<(int)(NH / 256), 256, 0, stream>>>(agg, zbf, (int)NH);
    rn_gemm64<true, false, false, true>
        <<<(N + 127) / 128, 256, 0, stream>>>(zbf, wC1[l], cb1[l], nullptr, nullptr, tconv, N);
    if (l == 2) {
      rn_gemm64<true, true, true, true>
          <<<(N + 127) / 128, 256, 0, stream>>>(tconv, wC2[l], cb2[l], hc, hn, hbf, N);
    } else {
      rn_gemm64<true, true, true, false>
          <<<(N + 127) / 128, 256, 0, stream>>>(tconv, wC2[l], cb2[l], hc, hn, nullptr, N);
    }
    float* tmp = hc; hc = hn; hn = tmp;
  }

  // Fused decoder
  rn_decoder<<<E / 128, 256, 110592, stream>>>(hbf, eeB, ei, wD1, wD2,
                                               db1, db2, dW3, db3,
                                               (float*)d_out, E);
}
